// LSTM_82523501625826
// MI455X (gfx1250) — compile-verified, matrix operand order checked by reading, not run on hardware
//
#include <hip/hip_runtime.h>
#include <hip/hip_bf16.h>

// ---------------------------------------------------------------------------
// LSTM (B=1024,T=512,I=2,H=256) + MLP head for MI455X (gfx1250):
//  - batch-partitioned persistent recurrence (no grid sync)
//  - v_wmma_f32_16x16x32_bf16 for h @ W_hh^T (f32 accumulate)
//  - W_hh K-slices staged global->LDS by the Tensor Data Mover (TENSORcnt),
//    double-buffered; fragments read via ds_load_b128 (single-base LDS
//    pointers so address-space inference keeps everything on the DS path)
// ---------------------------------------------------------------------------

typedef __attribute__((ext_vector_type(16))) __bf16 v16bf;
typedef __attribute__((ext_vector_type(8)))  float  v8f;
typedef __attribute__((ext_vector_type(4)))  unsigned int u32x4;
typedef __attribute__((ext_vector_type(8)))  int i32x8;
typedef __attribute__((ext_vector_type(4)))  int i32x4;

#define BATCH 1024
#define TLEN  512
#define HDIM  256
#define G4    1024
#define MB    32          // batch rows per workgroup
#define NWG   (BATCH/MB)  // 32 workgroups
#define KT    32          // K per WMMA slice
#define NKT   (HDIM/KT)   // 8 K-steps
#define WSTAGE_ELEMS (G4 * KT)                     // 32K bf16 = 64 KB / buffer
#define SMEM_BYTES   (2*WSTAGE_ELEMS*2 + MB*HDIM*2 + MB*2*4)

struct Frag { uint4 lo, hi; };
union FragU { Frag s; v16bf v; };

static __device__ __forceinline__ v16bf mkfrag(uint4 lo, uint4 hi) {
    FragU u; u.s.lo = lo; u.s.hi = hi; return u.v;
}

static __device__ __forceinline__ unsigned short f2bf(float f) {
    unsigned int u = __float_as_uint(f);
    u += 0x7FFFu + ((u >> 16) & 1u);          // round-to-nearest-even
    return (unsigned short)(u >> 16);
}

static __device__ __forceinline__ float fast_sig(float x) {
    // v_exp_f32 / v_rcp_f32 are TRANS ops -> co-execute with XDL WMMA
    return __builtin_amdgcn_rcpf(1.0f + __builtin_amdgcn_exp2f(-1.4426950408889634f * x));
}
static __device__ __forceinline__ float fast_tanh(float x) {
    return 2.0f * fast_sig(2.0f * x) - 1.0f;
}

// --- Tensor Data Mover: stage a [1024 x 32] bf16 tile of W_hh into LDS ------
#if __has_builtin(__builtin_amdgcn_tensor_load_to_lds)
#define USE_TDM 1
#endif

static __device__ __forceinline__ void tdm_load_wtile(
    const unsigned short* __restrict__ gsrc,  // tile start (row 0, col kt*32)
    unsigned short* ldst)                     // LDS destination buffer
{
#ifdef USE_TDM
    // D# group 0: count=1 | lds_addr | 57-bit global addr | type=2 (image)
    unsigned long long ga = (unsigned long long)(uintptr_t)gsrc;
    u32x4 g0;
    g0[0] = 1u;                                             // count=1, user mode
    g0[1] = (unsigned int)(uintptr_t)ldst;                  // LDS byte address
    g0[2] = (unsigned int)(ga & 0xFFFFFFFFull);             // global_addr[31:0]
    g0[3] = (unsigned int)((ga >> 32) & 0x1FFFFFFull) | (2u << 30); // [56:32]|type
    // D# group 1: wg_mask=0, data_size=1(2B); tensor 256x1024, tile 32x1024,
    //             dim0 stride = 256 elements
    i32x8 g1;
    g1[0] = 0x00010000;                 // data_size=1 (2 bytes)
    g1[1] = (int)(HDIM << 16);          // tensor_dim0[15:0] in bits[63:48]
    g1[2] = (int)(G4 << 16);            // tensor_dim0 hi | tensor_dim1[15:0]
    g1[3] = (int)(KT << 16);            // tensor_dim1 hi | tile_dim0=32
    g1[4] = G4;                         // tile_dim1=1024, tile_dim2=0
    g1[5] = HDIM;                       // tensor_dim0_stride[31:0] = 256
    g1[6] = 0;
    g1[7] = 0;
    i32x4 gz = {0, 0, 0, 0};
#if defined(__clang_major__) && (__clang_major__ >= 23)
    i32x8 gz8 = {0, 0, 0, 0, 0, 0, 0, 0};
    __builtin_amdgcn_tensor_load_to_lds(g0, g1, gz, gz, gz8, 0);
#else
    __builtin_amdgcn_tensor_load_to_lds(g0, g1, gz, gz, 0);
#endif
#else
    // fallback: cooperative per-lane copy (wave 0 only calls this)
    const int l = threadIdx.x & 31;
    for (int r = l; r < G4; r += 32) {
        uint4* d = (uint4*)&ldst[r * KT];
        const uint4* s = (const uint4*)&gsrc[(size_t)r * HDIM];
        d[0] = s[0]; d[1] = s[1]; d[2] = s[2]; d[3] = s[3];
    }
#endif
}

static __device__ __forceinline__ void tdm_wait_le1(void) {
#ifdef USE_TDM
    __builtin_amdgcn_s_wait_tensorcnt(1);
#endif
}
static __device__ __forceinline__ void tdm_wait_le0(void) {
#ifdef USE_TDM
    __builtin_amdgcn_s_wait_tensorcnt(0);
#endif
}

// --- prologue: W_hh f32 -> bf16 ---------------------------------------------
__global__ void convert_whh_kernel(const float* __restrict__ w,
                                   unsigned short* __restrict__ out, int n) {
    int i = blockIdx.x * blockDim.x + threadIdx.x;
    if (i < n) out[i] = f2bf(w[i]);
}

// --- main recurrent kernel ---------------------------------------------------
__global__ __launch_bounds__(256) void lstm_kernel(
    const float* __restrict__ x,            // [B, T, 2]
    const float* __restrict__ W_ih,         // [1024, 2]
    const float* __restrict__ b_ih,         // [1024]
    const float* __restrict__ b_hh,         // [1024]
    const unsigned short* __restrict__ whh, // [1024, 256] bf16
    float* __restrict__ h_out)              // [B, 256] f32
{
    // Single LDS base; every shared pointer is base+offset so clang keeps the
    // DS address space (avoids generic/FLAT fallback seen when selecting
    // between pointers stored in a private array).
    extern __shared__ __align__(16) unsigned short smem_us[];
    unsigned short* const wbase   = smem_us;                     // 2 x 64 KB
    unsigned short* const h_lds   = smem_us + 2 * WSTAGE_ELEMS;  // 16 KB bf16
    float*          const x_stage = (float*)(smem_us + 2 * WSTAGE_ELEMS + MB * HDIM);

    const int tid  = threadIdx.x;
    const int lane = tid & 31;
    const int wid  = tid >> 5;           // wave 0..7 -> h-columns [32w, 32w+32)
    const int row0 = blockIdx.x * MB;

    const int ncol = lane & 15;          // N within 16-wide tile
    const int hi   = lane >> 4;          // lane half
    const int koff = hi ? 8 : 0;         // K sub-offset (documented A layout)

    // zero initial hidden state
    for (int i = tid; i < MB * HDIM; i += 256) h_lds[i] = 0;

    // loop-invariant fragment offsets (elements)
    int aoff[2];                         // A: h_lds[(mt*16+ncol)*HDIM + koff]
#pragma unroll
    for (int mt = 0; mt < 2; ++mt) aoff[mt] = (mt * 16 + ncol) * HDIM + koff;
    int boff[4][2];                      // B: wbuf[n*KT + koff]
#pragma unroll
    for (int g = 0; g < 4; ++g)
#pragma unroll
        for (int nt = 0; nt < 2; ++nt)
            boff[g][nt] = (g * HDIM + wid * 32 + nt * 16 + ncol) * KT + koff;

    // per-lane constants: input-projection weights + bias for owned gate cols
    float wi0[4][2], wi1[4][2], bsum[4][2];
#pragma unroll
    for (int g = 0; g < 4; ++g)
#pragma unroll
        for (int nt = 0; nt < 2; ++nt) {
            int n = g * HDIM + wid * 32 + nt * 16 + ncol;
            wi0[g][nt]  = W_ih[n * 2 + 0];
            wi1[g][nt]  = W_ih[n * 2 + 1];
            bsum[g][nt] = b_ih[n] + b_hh[n];
        }

    // cell state in registers, tiled identically to the C-matrix layout
    v8f c[2][2];
    const v8f vzero = {0.f,0.f,0.f,0.f,0.f,0.f,0.f,0.f};
#pragma unroll
    for (int mt = 0; mt < 2; ++mt)
#pragma unroll
        for (int nt = 0; nt < 2; ++nt) c[mt][nt] = vzero;

    __syncthreads();

    for (int t = 0; t < TLEN; ++t) {
        // stage x[:, t, :] for this row-slice (64 floats)
        if (tid < MB * 2) {
            int r = tid >> 1, i = tid & 1;
            x_stage[tid] = x[((size_t)(row0 + r) * TLEN + t) * 2 + i];
        }
        // prime the double-buffered TDM pipeline (wave 0 issues)
        if (wid == 0) {
            tdm_load_wtile(whh + 0 * KT, wbase);
            tdm_load_wtile(whh + 1 * KT, wbase + WSTAGE_ELEMS);
        }

        v8f acc[4][2][2];
#pragma unroll
        for (int g = 0; g < 4; ++g)
#pragma unroll
            for (int mt = 0; mt < 2; ++mt)
#pragma unroll
                for (int nt = 0; nt < 2; ++nt) acc[g][mt][nt] = vzero;

        // ---- GEMM: gates[32, 4x32] += h[32,256] @ W_hh^T, TDM-fed ----
        // Barriers fence code motion each K-step, so a full unroll cannot
        // re-hoist loads across tiles (no spill risk) and yields 128 WMMAs.
#pragma unroll
        for (int kt = 0; kt < NKT; ++kt) {
            if (wid == 0) { if (kt == NKT - 1) tdm_wait_le0(); else tdm_wait_le1(); }
            __syncthreads();                     // staged tile kt is ready

            const unsigned short* wbuf = wbase + (kt & 1) * WSTAGE_ELEMS;
            v16bf afr[2];
#pragma unroll
            for (int mt = 0; mt < 2; ++mt) {
                const unsigned short* pa = h_lds + aoff[mt] + kt * KT;
                afr[mt] = mkfrag(*(const uint4*)pa, *(const uint4*)(pa + 16));
            }
#pragma unroll
            for (int g = 0; g < 4; ++g)
#pragma unroll
                for (int nt = 0; nt < 2; ++nt) {
                    const unsigned short* pb = wbuf + boff[g][nt];
                    v16bf bfr = mkfrag(*(const uint4*)pb, *(const uint4*)(pb + 16));
#pragma unroll
                    for (int mt = 0; mt < 2; ++mt)
                        acc[g][mt][nt] = __builtin_amdgcn_wmma_f32_16x16x32_bf16(
                            false, afr[mt], false, bfr, (short)0,
                            acc[g][mt][nt], false, false);
                }

            __syncthreads();                     // tile kt fully consumed
            if (wid == 0 && kt + 2 < NKT)        // refill this buffer
                tdm_load_wtile(whh + (kt + 2) * KT,
                               wbase + (kt & 1) * WSTAGE_ELEMS);
        }

        // ---- elementwise gate math + state update ----
#pragma unroll
        for (int mt = 0; mt < 2; ++mt) {
            float x0[8], x1[8];
#pragma unroll
            for (int v = 0; v < 8; ++v) {
                int rl = mt * 16 + hi * 8 + v;
                x0[v] = x_stage[rl * 2 + 0];
                x1[v] = x_stage[rl * 2 + 1];
            }
#pragma unroll
            for (int nt = 0; nt < 2; ++nt) {
#pragma unroll
                for (int v = 0; v < 8; ++v) {
                    float gi = acc[0][mt][nt][v] + x0[v]*wi0[0][nt] + x1[v]*wi1[0][nt] + bsum[0][nt];
                    float gf = acc[1][mt][nt][v] + x0[v]*wi0[1][nt] + x1[v]*wi1[1][nt] + bsum[1][nt];
                    float gg = acc[2][mt][nt][v] + x0[v]*wi0[2][nt] + x1[v]*wi1[2][nt] + bsum[2][nt];
                    float go = acc[3][mt][nt][v] + x0[v]*wi0[3][nt] + x1[v]*wi1[3][nt] + bsum[3][nt];
                    float iv = fast_sig(gi);
                    float fv = fast_sig(gf);
                    float gv = fast_tanh(gg);
                    float ov = fast_sig(go);
                    float cv = fv * c[mt][nt][v] + iv * gv;
                    c[mt][nt][v] = cv;
                    float hv = ov * fast_tanh(cv);
                    int rl  = mt * 16 + hi * 8 + v;
                    int col = wid * 32 + nt * 16 + ncol;
                    h_lds[rl * HDIM + col] = f2bf(hv);
                    if (t == TLEN - 1)
                        h_out[(size_t)(row0 + rl) * HDIM + col] = hv;
                }
            }
        }
        __syncthreads();   // h_lds/x_stage safe for next step
    }
}

// --- MLP head (0.1% of total FLOPs; simple LDS-tiled VALU) ------------------
__global__ __launch_bounds__(256) void mlp_layer_kernel(
    const float* __restrict__ in,  // [1024, 256]
    const float* __restrict__ W,   // [256, 256] (out = in @ W^T + b)
    const float* __restrict__ b,   // [256]
    float* __restrict__ out,       // [1024, 256]
    int do_relu)
{
    __shared__ float row[256];
    const int bb = blockIdx.x;
    const int j  = threadIdx.x;
    row[j] = in[bb * 256 + j];
    __syncthreads();
    float s = b[j];
    const float* wr = &W[j * 256];
#pragma unroll 8
    for (int k = 0; k < 256; ++k) s = fmaf(row[k], wr[k], s);
    if (do_relu) s = fmaxf(s, 0.0f);
    out[bb * 256 + j] = s;
}

__global__ void mlp_out_kernel(const float* __restrict__ in,   // [1024, 256]
                               const float* __restrict__ W3,   // [256]
                               const float* __restrict__ b3,   // [1]
                               float* __restrict__ out)        // [1024]
{
    int bb = blockIdx.x * blockDim.x + threadIdx.x;
    if (bb < BATCH) {
        float s = b3[0];
        const float* r = &in[bb * 256];
#pragma unroll 8
        for (int k = 0; k < 256; ++k) s = fmaf(r[k], W3[k], s);
        out[bb] = s;
    }
}

// ---------------------------------------------------------------------------
extern "C" void kernel_launch(void* const* d_in, const int* in_sizes, int n_in,
                              void* d_out, int out_size, void* d_ws, size_t ws_size,
                              hipStream_t stream) {
    const float* x    = (const float*)d_in[0];
    const float* W_ih = (const float*)d_in[1];
    const float* W_hh = (const float*)d_in[2];
    const float* b_ih = (const float*)d_in[3];
    const float* b_hh = (const float*)d_in[4];
    const float* W1   = (const float*)d_in[5];
    const float* b1   = (const float*)d_in[6];
    const float* W2   = (const float*)d_in[7];
    const float* b2   = (const float*)d_in[8];
    const float* W3   = (const float*)d_in[9];
    const float* b3   = (const float*)d_in[10];
    float* out = (float*)d_out;

    // workspace layout
    unsigned short* whh_bf = (unsigned short*)d_ws;                 // 512 KB
    float* hbuf = (float*)((char*)d_ws + (size_t)1024 * 256 * 2);   // 1 MB
    float* o1   = hbuf + (size_t)1024 * 256;                        // 1 MB
    float* o2   = o1   + (size_t)1024 * 256;                        // 1 MB

    convert_whh_kernel<<<(1024 * 256 + 255) / 256, 256, 0, stream>>>(
        W_hh, whh_bf, 1024 * 256);
    lstm_kernel<<<NWG, 256, SMEM_BYTES, stream>>>(x, W_ih, b_ih, b_hh, whh_bf, hbuf);
    mlp_layer_kernel<<<1024, 256, 0, stream>>>(hbuf, W1, b1, o1, 1);
    mlp_layer_kernel<<<1024, 256, 0, stream>>>(o1, W2, b2, o2, 1);
    mlp_out_kernel<<<(BATCH + 255) / 256, 256, 0, stream>>>(o2, W3, b3, out);
}
